// HierarchicalMoE_17368847745267
// MI455X (gfx1250) — compile-verified, matrix operand order checked by reading, not run on hardware
//
#include <hip/hip_runtime.h>

#define DI __device__ __forceinline__

// ---------------- problem constants ----------------
constexpr int B_ = 2, S_ = 1024, H_ = 1024, I_ = 4096;
constexpr int G_ = 4, E_ = 4, NE_ = 16;
constexpr int CAP_ = 1024;
constexpr int T_ = B_ * S_;       // 2048 tokens
constexpr int NSLOT_ = 2 * T_;    // 4096 routed slots

// ---------------- tiling ----------------
constexpr int BM = 128, BN = 128, KC = 32;
constexpr int LDT = 40;           // LDS row stride in bf16 elems (80B, 16B aligned)

typedef __attribute__((ext_vector_type(16))) __bf16       v16bf;
typedef __attribute__((ext_vector_type(8)))  float        v8f;
typedef __attribute__((ext_vector_type(4)))  unsigned int v4u;
typedef __attribute__((ext_vector_type(4)))  float        v4f;

union Frag { v4u u[2]; v16bf v; };

DI unsigned short f2bf(float f) {           // round-to-nearest-even fp32 -> bf16
  unsigned int u = __float_as_uint(f);
  u += 0x7FFFu + ((u >> 16) & 1u);
  return (unsigned short)(u >> 16);
}
DI unsigned int pack2(float a, float b) {
  return (unsigned int)f2bf(a) | ((unsigned int)f2bf(b) << 16);
}
DI v4u pack8(v4f lo, v4f hi) {
  v4u r = {pack2(lo[0], lo[1]), pack2(lo[2], lo[3]),
           pack2(hi[0], hi[1]), pack2(hi[2], hi[3])};
  return r;
}
DI v16bf load_frag(const unsigned short* p) {
  Frag f;
  const v4u* q = (const v4u*)p;
  f.u[0] = q[0]; f.u[1] = q[1];
  return f.v;
}
DI v8f wmma_bf16(v16bf a, v16bf b, v8f c) {
  return __builtin_amdgcn_wmma_f32_16x16x32_bf16(false, a, false, b, (short)0, c,
                                                 false, false);
}

// ---------------- kernel 1: fp32 -> bf16 token conversion ----------------
__global__ __launch_bounds__(256) void cvtx_kernel(const float* __restrict__ x,
                                                   unsigned short* __restrict__ xbf) {
  size_t i = ((size_t)blockIdx.x * 256 + threadIdx.x) * 4;   // 2048 blocks * 256 * 4
  v4f v = *(const v4f*)(x + i);
  unsigned int lo = pack2(v[0], v[1]);
  unsigned int hi = pack2(v[2], v[3]);
  unsigned int* o = (unsigned int*)(xbf + i);
  o[0] = lo; o[1] = hi;
}

// ---------------- kernel 2: hierarchical router ----------------
// one wave (32 lanes) per token; 4 waves per block
__global__ __launch_bounds__(128) void router_kernel(const float* __restrict__ x,
                                                     const float* __restrict__ gw,
                                                     const float* __restrict__ erw,
                                                     int* __restrict__ eid,
                                                     float* __restrict__ wgt) {
  const int lane = threadIdx.x & 31;
  const int t = blockIdx.x * 4 + (threadIdx.x >> 5);
  float acc[20];
#pragma unroll
  for (int i = 0; i < 20; ++i) acc[i] = 0.f;
  const float* xr = x + (size_t)t * H_;
  for (int h = lane; h < H_; h += 32) {
    float xv = xr[h];
#pragma unroll
    for (int g = 0; g < G_; ++g) acc[g] += xv * gw[g * H_ + h];
#pragma unroll
    for (int k = 0; k < NE_; ++k) acc[4 + k] += xv * erw[k * H_ + h];
  }
#pragma unroll
  for (int i = 0; i < 20; ++i)
#pragma unroll
    for (int off = 16; off > 0; off >>= 1) acc[i] += __shfl_xor(acc[i], off, 32);

  if (lane == 0) {
    // group softmax (fp32) + top-2 (first index wins on ties, as jax top_k)
    float gm = fmaxf(fmaxf(acc[0], acc[1]), fmaxf(acc[2], acc[3]));
    float ge[4], gs = 0.f;
#pragma unroll
    for (int g = 0; g < 4; ++g) { ge[g] = __expf(acc[g] - gm); gs += ge[g]; }
    float gp[4];
#pragma unroll
    for (int g = 0; g < 4; ++g) gp[g] = ge[g] / gs;
    int i0 = 0;
#pragma unroll
    for (int g = 1; g < 4; ++g) if (gp[g] > gp[i0]) i0 = g;
    int i1 = -1; float bv = -1e30f;
#pragma unroll
    for (int g = 0; g < 4; ++g) if (g != i0 && gp[g] > bv) { bv = gp[g]; i1 = g; }

    int gi[2] = {i0, i1};
#pragma unroll
    for (int s = 0; s < 2; ++s) {
      int g = gi[s];
      float l0 = acc[4 + g * 4 + 0], l1 = acc[4 + g * 4 + 1];
      float l2 = acc[4 + g * 4 + 2], l3 = acc[4 + g * 4 + 3];
      float m = fmaxf(fmaxf(l0, l1), fmaxf(l2, l3));
      float s0 = __expf(l0 - m), s1 = __expf(l1 - m);
      float s2 = __expf(l2 - m), s3 = __expf(l3 - m);
      float es = s0 + s1 + s2 + s3;
      int bi = 0; float bl = l0;
      if (l1 > bl) { bl = l1; bi = 1; }
      if (l2 > bl) { bl = l2; bi = 2; }
      if (l3 > bl) { bl = l3; bi = 3; }
      float prob = __expf(bl - m) / es;
      eid[2 * t + s] = g * E_ + bi;
      wgt[2 * t + s] = gp[g] * prob;
    }
  }
}

// ---------------- kernel 3: capacity dispatch (ordered per-expert scan) ------
__global__ __launch_bounds__(256) void dispatch_kernel(const int* __restrict__ eid,
                                                       int* __restrict__ token_of,
                                                       int* __restrict__ keep,
                                                       int* __restrict__ counts) {
  const int e = blockIdx.x;
  const int tid = threadIdx.x;
  __shared__ int scan[256];
  __shared__ int sbase;
  if (tid == 0) sbase = 0;
  __syncthreads();
  for (int c0 = 0; c0 < NSLOT_; c0 += 256) {
    int i = c0 + tid;
    int m = (eid[i] == e) ? 1 : 0;
    scan[tid] = m;
    __syncthreads();
    for (int off = 1; off < 256; off <<= 1) {
      int v = (tid >= off) ? scan[tid - off] : 0;
      __syncthreads();
      scan[tid] += v;
      __syncthreads();
    }
    int incl = scan[tid];
    int total = scan[255];
    int pos = sbase + incl - 1;
    if (m) {
      if (pos < CAP_) { token_of[e * CAP_ + pos] = i; keep[i] = 1; }
      else keep[i] = 0;
    }
    __syncthreads();
    if (tid == 0) sbase += total;
    __syncthreads();
  }
  if (tid == 0) counts[e] = (sbase < CAP_) ? sbase : CAP_;
}

// ---------------- kernel 4: grouped gate/up GEMM + SwiGLU (bf16 WMMA) --------
// grid: (I/BN, CAP/BM, NE), 256 threads = 8 waves, wave w owns 16x128 strip.
// Double-buffered LDS, register prefetch of the next K-chunk (raw fp32 weights
// held in VGPRs; bf16 conversion deferred to the LDS-store phase).
__global__ __launch_bounds__(256) void gemm1_kernel(const unsigned short* __restrict__ xbf,
                                                    const float* __restrict__ gate_w,
                                                    const float* __restrict__ up_w,
                                                    const int* __restrict__ token_of,
                                                    const int* __restrict__ counts,
                                                    unsigned short* __restrict__ act) {
  const int e = blockIdx.z;
  const int m0 = blockIdx.y * BM;
  const int n0 = blockIdx.x * BN;
  const int cnt = counts[e];
  if (m0 >= cnt) return;

  __shared__ unsigned short As[2][BM * LDT];
  __shared__ unsigned short Bg[2][BN * LDT];
  __shared__ unsigned short Bu[2][BN * LDT];
  __shared__ int s_tok[BM];

  const int tid = threadIdx.x;
  const int lane = tid & 31;
  const int wv = tid >> 5;
  const int l15 = lane & 15;
  const int lhalf = lane >> 4;

  if (tid < BM) {
    int rg = m0 + tid;
    s_tok[tid] = (rg < cnt) ? token_of[e * CAP_ + rg] : -1;
  }
  __syncthreads();

  const v8f zero8 = {0.f, 0.f, 0.f, 0.f, 0.f, 0.f, 0.f, 0.f};
  v8f accg[8], accu[8];
#pragma unroll
  for (int n = 0; n < 8; ++n) { accg[n] = zero8; accu[n] = zero8; }

  const int row = tid >> 1;        // 0..127 (A row / B row)
  const int koff = (tid & 1) * 16; // half of 32-wide K slab
  const size_t wbase = ((size_t)e * I_ + (n0 + row)) * H_;
  const int slot = s_tok[row];
  const unsigned short* aptr =
      (slot >= 0) ? (xbf + ((size_t)(slot >> 1)) * H_ + koff) : nullptr;

  // staging registers
  v4u a0 = {0u, 0u, 0u, 0u}, a1 = {0u, 0u, 0u, 0u};
  v4f g0, g1, g2, g3, u0, u1, u2, u3;

  // prologue: fetch K-chunk 0
  if (aptr) { const v4u* p = (const v4u*)aptr; a0 = p[0]; a1 = p[1]; }
  {
    const v4f* pg = (const v4f*)(gate_w + wbase + koff);
    const v4f* pu = (const v4f*)(up_w + wbase + koff);
    g0 = pg[0]; g1 = pg[1]; g2 = pg[2]; g3 = pg[3];
    u0 = pu[0]; u1 = pu[1]; u2 = pu[2]; u3 = pu[3];
  }

  int buf = 0;
  for (int k0 = 0; k0 < H_; k0 += KC, buf ^= 1) {
    // store staged chunk into LDS[buf] (convert weights fp32 -> bf16 here)
    {
      v4u* da = (v4u*)(&As[buf][row * LDT + koff]);
      da[0] = a0; da[1] = a1;
      v4u* dg = (v4u*)(&Bg[buf][row * LDT + koff]);
      dg[0] = pack8(g0, g1); dg[1] = pack8(g2, g3);
      v4u* du = (v4u*)(&Bu[buf][row * LDT + koff]);
      du[0] = pack8(u0, u1); du[1] = pack8(u2, u3);
    }
    __syncthreads();

    // prefetch next K-chunk while WMMAs run out of LDS[buf]
    const int kn = k0 + KC;
    if (kn < H_) {
      if (aptr) {
        const v4u* p = (const v4u*)(aptr + kn);
        a0 = p[0]; a1 = p[1];
      }
      const v4f* pg = (const v4f*)(gate_w + wbase + kn + koff);
      const v4f* pu = (const v4f*)(up_w + wbase + kn + koff);
      g0 = pg[0]; g1 = pg[1]; g2 = pg[2]; g3 = pg[3];
      u0 = pu[0]; u1 = pu[1]; u2 = pu[2]; u3 = pu[3];
    }

    v16bf fa = load_frag(&As[buf][(wv * 16 + l15) * LDT + lhalf * 16]);
#pragma unroll
    for (int n = 0; n < 8; ++n) {
      v16bf fg = load_frag(&Bg[buf][(n * 16 + l15) * LDT + lhalf * 16]);
      accg[n] = wmma_bf16(fa, fg, accg[n]);
      v16bf fu = load_frag(&Bu[buf][(n * 16 + l15) * LDT + lhalf * 16]);
      accu[n] = wmma_bf16(fa, fu, accu[n]);
    }
  }

  // epilogue: act = silu(gate) * up, bf16
  const int rbase = m0 + wv * 16 + ((lane >= 16) ? 8 : 0);
  const int cbase = n0 + l15;
#pragma unroll
  for (int n = 0; n < 8; ++n) {
#pragma unroll
    for (int r = 0; r < 8; ++r) {
      float g = accg[n][r], u = accu[n][r];
      float v = (g / (1.f + __expf(-g))) * u;
      act[((size_t)e * CAP_ + (rbase + r)) * I_ + (cbase + n * 16)] = f2bf(v);
    }
  }
}

// ---------------- kernel 5: grouped down GEMM (bf16 WMMA) --------------------
// grid: (H/BN, CAP/BM, NE); same double-buffered pipeline, K = I = 4096.
__global__ __launch_bounds__(256) void gemm2_kernel(const unsigned short* __restrict__ act,
                                                    const float* __restrict__ down_w,
                                                    const int* __restrict__ token_of,
                                                    const int* __restrict__ counts,
                                                    float* __restrict__ yslot) {
  const int e = blockIdx.z;
  const int m0 = blockIdx.y * BM;
  const int n0 = blockIdx.x * BN;
  const int cnt = counts[e];
  if (m0 >= cnt) return;

  __shared__ unsigned short As[2][BM * LDT];
  __shared__ unsigned short Bd[2][BN * LDT];
  __shared__ int s_tok[BM];

  const int tid = threadIdx.x;
  const int lane = tid & 31;
  const int wv = tid >> 5;
  const int l15 = lane & 15;
  const int lhalf = lane >> 4;

  if (tid < BM) {
    int rg = m0 + tid;
    s_tok[tid] = (rg < cnt) ? token_of[e * CAP_ + rg] : -1;
  }
  __syncthreads();

  const v8f zero8 = {0.f, 0.f, 0.f, 0.f, 0.f, 0.f, 0.f, 0.f};
  v8f acc[8];
#pragma unroll
  for (int n = 0; n < 8; ++n) acc[n] = zero8;

  const int row = tid >> 1;
  const int koff = (tid & 1) * 16;
  const unsigned short* aptr = act + ((size_t)e * CAP_ + (m0 + row)) * I_ + koff;
  const float* wptr = down_w + ((size_t)e * H_ + (n0 + row)) * I_ + koff;

  v4u a0, a1;
  v4f d0, d1, d2, d3;

  // prologue: fetch K-chunk 0
  { const v4u* p = (const v4u*)aptr; a0 = p[0]; a1 = p[1]; }
  { const v4f* p = (const v4f*)wptr; d0 = p[0]; d1 = p[1]; d2 = p[2]; d3 = p[3]; }

  int buf = 0;
  for (int k0 = 0; k0 < I_; k0 += KC, buf ^= 1) {
    {
      v4u* da = (v4u*)(&As[buf][row * LDT + koff]);
      da[0] = a0; da[1] = a1;
      v4u* dd = (v4u*)(&Bd[buf][row * LDT + koff]);
      dd[0] = pack8(d0, d1); dd[1] = pack8(d2, d3);
    }
    __syncthreads();

    const int kn = k0 + KC;
    if (kn < I_) {
      const v4u* p = (const v4u*)(aptr + kn);
      a0 = p[0]; a1 = p[1];
      const v4f* q = (const v4f*)(wptr + kn);
      d0 = q[0]; d1 = q[1]; d2 = q[2]; d3 = q[3];
    }

    v16bf fa = load_frag(&As[buf][(wv * 16 + l15) * LDT + lhalf * 16]);
#pragma unroll
    for (int n = 0; n < 8; ++n) {
      v16bf fb = load_frag(&Bd[buf][(n * 16 + l15) * LDT + lhalf * 16]);
      acc[n] = wmma_bf16(fa, fb, acc[n]);
    }
  }

  // epilogue: scatter rows to slot-indexed output buffer
  const int rloc = wv * 16 + ((lane >= 16) ? 8 : 0);
#pragma unroll
  for (int n = 0; n < 8; ++n) {
#pragma unroll
    for (int r = 0; r < 8; ++r) {
      int rr = rloc + r;
      int rg = m0 + rr;
      if (rg < cnt) {
        int slot = s_tok[rr];
        yslot[(size_t)slot * H_ + n0 + n * 16 + l15] = acc[n][r];
      }
    }
  }
}

// ---------------- kernel 6: weighted combine -------------------------------
__global__ __launch_bounds__(256) void combine_kernel(const float* __restrict__ yslot,
                                                      const float* __restrict__ wgt,
                                                      const int* __restrict__ keep,
                                                      float* __restrict__ out) {
  const int t = blockIdx.x;
  const int h = blockIdx.y * 256 + threadIdx.x;
  const int s0 = 2 * t, s1 = s0 + 1;
  float v = 0.f;
  if (keep[s0]) v += wgt[s0] * yslot[(size_t)s0 * H_ + h];
  if (keep[s1]) v += wgt[s1] * yslot[(size_t)s1 * H_ + h];
  out[(size_t)t * H_ + h] = v;
  if (t == 0 && blockIdx.y == 0 && threadIdx.x == 0)
    out[(size_t)T_ * H_] = 0.f;  // aux_loss
}

// ---------------- host-side launch -----------------------------------------
extern "C" void kernel_launch(void* const* d_in, const int* in_sizes, int n_in,
                              void* d_out, int out_size, void* d_ws, size_t ws_size,
                              hipStream_t stream) {
  const float* x   = (const float*)d_in[0];  // [B,S,H]
  const float* gw  = (const float*)d_in[1];  // [G,H]
  const float* erw = (const float*)d_in[2];  // [G,E,H]
  const float* Wg  = (const float*)d_in[3];  // [NE,I,H]
  const float* Wu  = (const float*)d_in[4];  // [NE,I,H]
  const float* Wd  = (const float*)d_in[5];  // [NE,H,I]
  float* out = (float*)d_out;

  char* ws = (char*)d_ws;
  // workspace layout (bytes)
  constexpr size_t o_eid  = 0;                                   // 4096 * 4
  constexpr size_t o_wgt  = o_eid + (size_t)NSLOT_ * 4;          // 4096 * 4
  constexpr size_t o_keep = o_wgt + (size_t)NSLOT_ * 4;          // 4096 * 4
  constexpr size_t o_tok  = o_keep + (size_t)NSLOT_ * 4;         // 16*1024 * 4
  constexpr size_t o_cnt  = o_tok + (size_t)NE_ * CAP_ * 4;      // 16 * 4 (pad)
  constexpr size_t o_xbf  = 131072;                              // T*H bf16 = 4MB
  constexpr size_t o_act  = 8ull * 1024 * 1024;                  // NE*CAP*I bf16 = 128MB
  constexpr size_t o_ysl  = o_act + (size_t)NE_ * CAP_ * I_ * 2; // NSLOT*H f32 = 16MB

  int*   eid      = (int*)(ws + o_eid);
  float* wgt      = (float*)(ws + o_wgt);
  int*   keep     = (int*)(ws + o_keep);
  int*   token_of = (int*)(ws + o_tok);
  int*   counts   = (int*)(ws + o_cnt);
  unsigned short* xbf = (unsigned short*)(ws + o_xbf);
  unsigned short* act = (unsigned short*)(ws + o_act);
  float* yslot    = (float*)(ws + o_ysl);

  (void)in_sizes; (void)n_in; (void)out_size; (void)ws_size;

  cvtx_kernel<<<(T_ * H_) / (256 * 4), 256, 0, stream>>>(x, xbf);
  router_kernel<<<T_ / 4, 128, 0, stream>>>(x, gw, erw, eid, wgt);
  dispatch_kernel<<<NE_, 256, 0, stream>>>(eid, token_of, keep, counts);
  gemm1_kernel<<<dim3(I_ / BN, CAP_ / BM, NE_), 256, 0, stream>>>(
      xbf, Wg, Wu, token_of, counts, act);
  gemm2_kernel<<<dim3(H_ / BN, CAP_ / BM, NE_), 256, 0, stream>>>(
      act, Wd, token_of, counts, yslot);
  combine_kernel<<<dim3(T_, H_ / 256), 256, 0, stream>>>(yslot, wgt, keep, out);
}
